// GATv2_20306605375762
// MI455X (gfx1250) — matrix operand order repaired and checked by reading, not compiled, not run on hardware
//
#include <hip/hip_runtime.h>
#include <math.h>

// ---------------------------------------------------------------------------
// GATv2 (2 layers, H=4, C=32, F=128) + mean-pool + MLP head for gfx1250.
// Dense node transforms: v_wmma_f32_16x16x32_bf16 with register-layout
// pre-packed B fragments (b128 loads) and LDS-staged A tiles.
// Edge softmax/aggregate: wave-per-edge coalesced gathers + native atomics.
// ---------------------------------------------------------------------------

typedef __bf16 bf16_t;
typedef __attribute__((ext_vector_type(16))) __bf16 v16bf;
typedef __attribute__((ext_vector_type(8)))  __bf16 v8bf;
typedef __attribute__((ext_vector_type(8)))  float  v8f;

#define N_NODES 50000
#define N_EDGES 800000
#define N_ETOT  850000   // + self loops
#define NHEAD   4
#define NCH     128      // H*C
#define NGRAPH  64
#define NEG_SLOPE 0.2f

// ---------------- conversion kernels ----------------

__global__ void k_cvt_x(const float* __restrict__ x, bf16_t* __restrict__ xb, int n) {
  int i = blockIdx.x * blockDim.x + threadIdx.x;
  if (i < n) xb[i] = (bf16_t)x[i];
}

// W is [O=128][K=128] f32 row-major. Pack into WMMA B-fragment register
// layout: Wp[jt][kb][lane][elem], elem 0..15 contiguous (32B per lane) so the
// GEMM loads each fragment with two global_load_b128.
// ISA 7.12.2 B 32x16 bf16: col = jt*16 + lane%16; lanes<16 hold K {0..7,16..23},
// lanes>=16 hold K {8..15,24..31} of K-block kb.
__global__ void k_pack_wB(const float* __restrict__ W, bf16_t* __restrict__ Wp) {
  int i = blockIdx.x * blockDim.x + threadIdx.x;
  if (i >= 128 * 128) return;
  int elem =  i        & 15;
  int lane = (i >> 4)  & 31;
  int kb   = (i >> 9)  & 3;
  int jt   = (i >> 11) & 7;
  int o = jt * 16 + (lane & 15);
  int k = kb * 32 + ((lane < 16) ? 0 : 8) + (elem & 7) + ((elem & 8) << 1);
  Wp[i] = (bf16_t)W[o * 128 + k];
}

// ---------------- WMMA GEMM: XL = Xb @ Wl.T, XR = Xb @ Wr.T (+bias) --------
// 256 threads = 8 waves; 64 rows of X staged in LDS (16KB). Wave j owns
// output column tile j and keeps BOTH weight matrices' B fragments for its
// columns resident in registers (8 x v16bf); each A fragment feeds two WMMAs.
__global__ __launch_bounds__(256) void k_gemm_xlxr(
    const bf16_t* __restrict__ Xb,    // [N,128] bf16
    const bf16_t* __restrict__ WlP,   // packed fragments
    const bf16_t* __restrict__ WrP,
    const float*  __restrict__ bl, const float* __restrict__ br,
    float* __restrict__ XL, float* __restrict__ XR)
{
  __shared__ __align__(16) bf16_t sX[64 * 128];   // 16 KB
  const int wave = threadIdx.x >> 5;              // 0..7 = column tile
  const int lane = threadIdx.x & 31;
  const int row0 = blockIdx.x * 64;

  // cooperative tile load: 4 iters x 256 threads x 8 bf16 (16B) = 64x128
#pragma unroll
  for (int it = 0; it < 4; ++it) {
    int off = (it * 256 + (int)threadIdx.x) * 8;  // element offset
    int r = off >> 7, c = off & 127;
    int grow = row0 + r;
    if (grow >= N_NODES) grow = N_NODES - 1;      // clamped (stores are guarded)
    *reinterpret_cast<uint4*>(&sX[off]) =
        *reinterpret_cast<const uint4*>(Xb + (size_t)grow * 128 + c);
  }
  __syncthreads();

  // resident B fragments: [kb] for Wl and Wr (32B contiguous per lane)
  v16bf bfl[4], bfr[4];
  {
    const size_t base = (((size_t)wave * 4) * 32 + (size_t)lane) * 16;
#pragma unroll
    for (int kb = 0; kb < 4; ++kb) {
      bfl[kb] = *reinterpret_cast<const v16bf*>(WlP + base + (size_t)kb * 32 * 16);
      bfr[kb] = *reinterpret_cast<const v16bf*>(WrP + base + (size_t)kb * 32 * 16);
    }
  }

  const int   mlane = lane & 15;
  const int   col   = wave * 16 + mlane;
  const float bvl   = bl[col];
  const float bvr   = br[col];
  const int   rbase = (lane < 16) ? 0 : 8;

#pragma unroll
  for (int rt = 0; rt < 4; ++rt) {
    v8f accl = {}, accr = {};
#pragma unroll
    for (int kb = 0; kb < 4; ++kb) {
      // A 16x32 bf16 fragment (ISA 7.12.2): row = lane%16; lanes<16 hold
      // K {0..7,16..23}, lanes>=16 hold K {8..15,24..31} -> two ds_load_b128
      const int kbase = kb * 32 + ((lane < 16) ? 0 : 8);
      const bf16_t* rp = &sX[(rt * 16 + mlane) * 128 + kbase];
      v8bf alo = *reinterpret_cast<const v8bf*>(rp);
      v8bf ahi = *reinterpret_cast<const v8bf*>(rp + 16);
      v16bf a;
#pragma unroll
      for (int q = 0; q < 8; ++q) { a[q] = alo[q]; a[q + 8] = ahi[q]; }
      accl = __builtin_amdgcn_wmma_f32_16x16x32_bf16(
          false, a, false, bfl[kb], (short)0, accl, false, false);
      accr = __builtin_amdgcn_wmma_f32_16x16x32_bf16(
          false, a, false, bfr[kb], (short)0, accr, false, false);
    }
    // D layout: VGPR r -> M = rbase + r, N = lane%16
#pragma unroll
    for (int r = 0; r < 8; ++r) {
      const int row = row0 + rt * 16 + rbase + r;
      if (row < N_NODES) {
        XL[(size_t)row * 128 + col] = accl[r] + bvl;
        XR[(size_t)row * 128 + col] = accr[r] + bvr;
      }
    }
  }
}

// ---------------- per-layer init ----------------

__global__ void k_init(float* __restrict__ amax, float* __restrict__ denom,
                       float* __restrict__ agg) {
  int i = blockIdx.x * blockDim.x + threadIdx.x;
  if (i < N_NODES * NHEAD) { amax[i] = -__builtin_inff(); denom[i] = 0.0f; }
  if (i < N_NODES * NCH)   agg[i] = 0.0f;
}

// order-preserving float max via integer atomics (global_atomic_max_i32 /
// global_atomic_min_u32; no CAS loop)
__device__ inline void atomicMaxF(float* addr, float v) {
  int iv = __float_as_int(v);
  if (iv >= 0) atomicMax((int*)addr, iv);
  else         atomicMin((unsigned int*)addr, (unsigned int)iv);
}

// ---------------- edge pass 1: logits + segment max --------------------
// one wave per edge; lane owns 4 channels (float4), head = lane/8
__global__ __launch_bounds__(256) void k_edge_logits(
    const int* __restrict__ ei, const float* __restrict__ XL,
    const float* __restrict__ XR, const float* __restrict__ att,
    float* __restrict__ logits, float* __restrict__ amax)
{
  const int e = blockIdx.x * 8 + (threadIdx.x >> 5);
  if (e >= N_ETOT) return;
  const int lane = threadIdx.x & 31;
  int src, dst;
  if (e < N_EDGES) { src = ei[e]; dst = ei[N_EDGES + e]; }
  else             { src = dst = e - N_EDGES; }

  const float4 xl = *reinterpret_cast<const float4*>(XL + (size_t)src * NCH + lane * 4);
  const float4 xr = *reinterpret_cast<const float4*>(XR + (size_t)dst * NCH + lane * 4);
  const float4 av = *reinterpret_cast<const float4*>(att + lane * 4);

  float p = 0.0f, m;
  m = xl.x + xr.x; m = (m > 0.0f) ? m : NEG_SLOPE * m; p += av.x * m;
  m = xl.y + xr.y; m = (m > 0.0f) ? m : NEG_SLOPE * m; p += av.y * m;
  m = xl.z + xr.z; m = (m > 0.0f) ? m : NEG_SLOPE * m; p += av.z * m;
  m = xl.w + xr.w; m = (m > 0.0f) ? m : NEG_SLOPE * m; p += av.w * m;

  // reduce within 8-lane head groups
  p += __shfl_xor(p, 1, 32);
  p += __shfl_xor(p, 2, 32);
  p += __shfl_xor(p, 4, 32);

  if ((lane & 7) == 0) {
    const int h = lane >> 3;
    logits[(size_t)e * NHEAD + h] = p;
    atomicMaxF(&amax[dst * NHEAD + h], p);
  }
}

// ---------------- edge pass 2: exp + segment sum -----------------------

__global__ void k_edge_exp(const int* __restrict__ ei, float* __restrict__ logits,
                           const float* __restrict__ amax, float* __restrict__ denom)
{
  const int idx = blockIdx.x * blockDim.x + threadIdx.x;
  if (idx >= N_ETOT * NHEAD) return;
  const int e = idx >> 2, h = idx & 3;
  const int dst = (e < N_EDGES) ? ei[N_EDGES + e] : (e - N_EDGES);
  const float ex = __expf(logits[idx] - amax[dst * NHEAD + h]);
  logits[idx] = ex;                         // reuse buffer as ex
  atomicAdd(&denom[dst * NHEAD + h], ex);
}

// ---------------- edge pass 3: weighted aggregate ----------------------

__global__ __launch_bounds__(256) void k_edge_aggr(
    const int* __restrict__ ei, const float* __restrict__ XL,
    const float* __restrict__ ex, const float* __restrict__ denom,
    float* __restrict__ agg)
{
  const int e = blockIdx.x * 8 + (threadIdx.x >> 5);
  if (e >= N_ETOT) return;
  const int lane = threadIdx.x & 31;
  int src, dst;
  if (e < N_EDGES) { src = ei[e]; dst = ei[N_EDGES + e]; }
  else             { src = dst = e - N_EDGES; }

  const int h = lane >> 3;
  const float alpha = ex[(size_t)e * NHEAD + h] / denom[dst * NHEAD + h];
  const float4 xl = *reinterpret_cast<const float4*>(XL + (size_t)src * NCH + lane * 4);
  float* o = agg + (size_t)dst * NCH + lane * 4;
  atomicAdd(o + 0, xl.x * alpha);
  atomicAdd(o + 1, xl.y * alpha);
  atomicAdd(o + 2, xl.z * alpha);
  atomicAdd(o + 3, xl.w * alpha);
}

// ---------------- bias + relu (in place) + bf16 copy for next layer ----

__global__ void k_bias_relu_cvt(float* __restrict__ agg, const float* __restrict__ bias,
                                bf16_t* __restrict__ xb, int n)
{
  const int i = blockIdx.x * blockDim.x + threadIdx.x;
  if (i >= n) return;
  float v = agg[i] + bias[i & 127];
  v = (v > 0.0f) ? v : 0.0f;
  agg[i] = v;
  xb[i] = (bf16_t)v;
}

// ---------------- pooling + classifier --------------------------------

__global__ void k_pool_init(float* __restrict__ gsum, float* __restrict__ gcnt) {
  const int i = blockIdx.x * blockDim.x + threadIdx.x;
  if (i < NGRAPH * NCH) gsum[i] = 0.0f;
  if (i < NGRAPH)       gcnt[i] = 0.0f;
}

__global__ void k_pool(const float* __restrict__ h, const int* __restrict__ batch,
                       float* __restrict__ gsum, float* __restrict__ gcnt)
{
  const int i = blockIdx.x * blockDim.x + threadIdx.x;
  if (i >= N_NODES * NCH) return;
  const int node = i >> 7, c = i & 127;
  const int g = batch[node];
  atomicAdd(&gsum[g * NCH + c], h[i]);
  if (c == 0) atomicAdd(&gcnt[g], 1.0f);
}

__global__ void k_classifier(const float* __restrict__ gsum, const float* __restrict__ gcnt,
                             const float* __restrict__ Wc1, const float* __restrict__ bc1,
                             const float* __restrict__ Wc2, const float* __restrict__ bc2,
                             float* __restrict__ out)
{
  const int g = threadIdx.x;
  if (g >= NGRAPH) return;
  float cnt = gcnt[g]; cnt = (cnt > 1.0f) ? cnt : 1.0f;
  const float inv = 1.0f / cnt;
  float z[8];
#pragma unroll
  for (int j = 0; j < 8; ++j) {
    float s = bc1[j];
    for (int k = 0; k < NCH; ++k) s += Wc1[j * NCH + k] * gsum[g * NCH + k] * inv;
    z[j] = (s > 0.0f) ? s : 0.0f;
  }
  float s2 = bc2[0];
#pragma unroll
  for (int j = 0; j < 8; ++j) s2 += Wc2[j] * z[j];
  out[g] = s2;
}

// ---------------------------------------------------------------------------

extern "C" void kernel_launch(void* const* d_in, const int* in_sizes, int n_in,
                              void* d_out, int out_size, void* d_ws, size_t ws_size,
                              hipStream_t stream) {
  (void)in_sizes; (void)n_in; (void)out_size; (void)ws_size;
  const float* x     = (const float*)d_in[0];
  const int*   ei    = (const int*)d_in[1];
  const int*   batch = (const int*)d_in[2];
  const float* Wl0 = (const float*)d_in[3];  const float* bl0 = (const float*)d_in[4];
  const float* Wr0 = (const float*)d_in[5];  const float* br0 = (const float*)d_in[6];
  const float* att0= (const float*)d_in[7];  const float* bias0=(const float*)d_in[8];
  const float* Wl1 = (const float*)d_in[9];  const float* bl1 = (const float*)d_in[10];
  const float* Wr1 = (const float*)d_in[11]; const float* br1 = (const float*)d_in[12];
  const float* att1= (const float*)d_in[13]; const float* bias1=(const float*)d_in[14];
  const float* Wc1 = (const float*)d_in[15]; const float* bc1 = (const float*)d_in[16];
  const float* Wc2 = (const float*)d_in[17]; const float* bc2 = (const float*)d_in[18];
  float* out = (float*)d_out;

  // scratch carving (256B aligned)
  char* p = (char*)d_ws;
  auto carve = [&](size_t bytes) {
    char* r = p; p += (bytes + 255) & ~(size_t)255; return r;
  };
  bf16_t* xbf   = (bf16_t*)carve((size_t)N_NODES * NCH * 2);
  bf16_t* wl0b  = (bf16_t*)carve(128 * 128 * 2);
  bf16_t* wr0b  = (bf16_t*)carve(128 * 128 * 2);
  bf16_t* wl1b  = (bf16_t*)carve(128 * 128 * 2);
  bf16_t* wr1b  = (bf16_t*)carve(128 * 128 * 2);
  float*  xl    = (float*)carve((size_t)N_NODES * NCH * 4);
  float*  xr    = (float*)carve((size_t)N_NODES * NCH * 4);
  float*  logits= (float*)carve((size_t)N_ETOT * NHEAD * 4);
  float*  amax  = (float*)carve((size_t)N_NODES * NHEAD * 4);
  float*  denom = (float*)carve((size_t)N_NODES * NHEAD * 4);
  float*  agg   = (float*)carve((size_t)N_NODES * NCH * 4);
  float*  gsum  = (float*)carve(NGRAPH * NCH * 4);
  float*  gcnt  = (float*)carve(NGRAPH * 4);

  const int nElems = N_NODES * NCH;                 // 6.4M
  const int gElem  = (nElems + 255) / 256;
  const int gEdgeW = (N_ETOT + 7) / 8;              // wave-per-edge, 8 waves/block
  const int gEdgeT = (N_ETOT * NHEAD + 255) / 256;
  const int gGemm  = (N_NODES + 63) / 64;           // 782

  // one-time conversions / weight packing
  k_cvt_x<<<gElem, 256, 0, stream>>>(x, xbf, nElems);
  k_pack_wB<<<64, 256, 0, stream>>>(Wl0, wl0b);
  k_pack_wB<<<64, 256, 0, stream>>>(Wr0, wr0b);
  k_pack_wB<<<64, 256, 0, stream>>>(Wl1, wl1b);
  k_pack_wB<<<64, 256, 0, stream>>>(Wr1, wr1b);

  // ---- layer 0 ----
  k_gemm_xlxr<<<gGemm, 256, 0, stream>>>(xbf, wl0b, wr0b, bl0, br0, xl, xr);
  k_init<<<gElem, 256, 0, stream>>>(amax, denom, agg);
  k_edge_logits<<<gEdgeW, 256, 0, stream>>>(ei, xl, xr, att0, logits, amax);
  k_edge_exp<<<gEdgeT, 256, 0, stream>>>(ei, logits, amax, denom);
  k_edge_aggr<<<gEdgeW, 256, 0, stream>>>(ei, xl, logits, denom, agg);
  k_bias_relu_cvt<<<gElem, 256, 0, stream>>>(agg, bias0, xbf, nElems);

  // ---- layer 1 ----
  k_gemm_xlxr<<<gGemm, 256, 0, stream>>>(xbf, wl1b, wr1b, bl1, br1, xl, xr);
  k_init<<<gElem, 256, 0, stream>>>(amax, denom, agg);
  k_edge_logits<<<gEdgeW, 256, 0, stream>>>(ei, xl, xr, att1, logits, amax);
  k_edge_exp<<<gEdgeT, 256, 0, stream>>>(ei, logits, amax, denom);
  k_edge_aggr<<<gEdgeW, 256, 0, stream>>>(ei, xl, logits, denom, agg);
  k_bias_relu_cvt<<<gElem, 256, 0, stream>>>(agg, bias1, xbf, nElems);

  // ---- pool + classifier ----
  k_pool_init<<<(NGRAPH * NCH + 255) / 256, 256, 0, stream>>>(gsum, gcnt);
  k_pool<<<gElem, 256, 0, stream>>>(agg, batch, gsum, gcnt);
  k_classifier<<<1, 64, 0, stream>>>(gsum, gcnt, Wc1, bc1, Wc2, bc2, out);
}
